// ExactLookupMerger_70265664962673
// MI455X (gfx1250) — matrix operand order repaired and checked by reading, not compiled, not run on hardware
//
#include <hip/hip_runtime.h>
#include <hip/hip_bf16.h>
#include <cstdint>
#include <cstddef>

// ---------------------------------------------------------------------------
// Problem dimensions (fixed by the reference)
// ---------------------------------------------------------------------------
#define BB      8192
#define IN_DIM  2048
#define H_DIM   8192
#define OUT_DIM 2048
#define CB_DIM  256

typedef __bf16 bf16;
typedef __attribute__((ext_vector_type(8)))  bf16  v8bf;
typedef __attribute__((ext_vector_type(16))) bf16  v16bf;
typedef __attribute__((ext_vector_type(8)))  float v8f;

// ---------------------------------------------------------------------------
// CDNA5 helpers
// ---------------------------------------------------------------------------

// Low 32 bits of a generic pointer to LDS == LDS byte offset (ISA 10.2 aperture:
// the LDS address is the low 32 address bits).
__device__ __forceinline__ uint32_t lds_addr_of(const void* p) {
    return (uint32_t)(uintptr_t)p;
}

// ASYNCcnt-tracked global->LDS copy: 16 bytes per lane, no VGPR staging.
// GVS mode: mem = SGPR64 + VGPR32-offset; LDS dest = per-lane VGPR byte addr.
__device__ __forceinline__ void async_copy_b128(uint32_t lds_byte_addr,
                                                uint32_t glb_byte_off,
                                                const void* sbase) {
    asm volatile("global_load_async_to_lds_b128 %0, %1, %2"
                 :: "v"(lds_byte_addr), "v"(glb_byte_off), "s"(sbase)
                 : "memory");
}

template <int N>
__device__ __forceinline__ void wait_asynccnt() {
    asm volatile("s_wait_asynccnt %0" :: "n"(N) : "memory");
}

// tanh via 2 TRANS ops (v_exp_f32 + v_rcp_f32), no IEEE divide chain.
__device__ __forceinline__ float fast_tanh(float y) {
    float ay = __builtin_fabsf(y);
    float e  = __builtin_amdgcn_exp2f(ay * -2.8853900818f);   // exp(-2|y|)
    float r  = __builtin_amdgcn_rcpf(1.0f + e);
    float t  = (1.0f - e) * r;
    return __builtin_copysignf(t, y);
}

// ---------------------------------------------------------------------------
// Prep kernels
// ---------------------------------------------------------------------------

__global__ __launch_bounds__(256) void f2bf_kernel(const float* __restrict__ in,
                                                   bf16* __restrict__ out, size_t n) {
    size_t i = (size_t)blockIdx.x * blockDim.x + threadIdx.x;
    if (i < n) out[i] = (bf16)in[i];
}

// Per hidden unit: rho = sigmoid(rho_raw); c = softplus(c_raw);
// q = (1-rho)*c ; ic = 1/c   (so epilogue is rho*v + q*tanh(v*ic))
__global__ __launch_bounds__(256) void act_params_kernel(const float* __restrict__ craw,
                                                         const float* __restrict__ rraw,
                                                         float* __restrict__ rhoA,
                                                         float* __restrict__ qA,
                                                         float* __restrict__ icA, int n) {
    int i = blockIdx.x * blockDim.x + threadIdx.x;
    if (i < n) {
        float xc = craw[i];
        float c  = (xc > 20.0f) ? xc : log1pf(expf(xc));
        float rho = 1.0f / (1.0f + expf(-rraw[i]));
        rhoA[i] = rho;
        qA[i]   = (1.0f - rho) * c;
        icA[i]  = 1.0f / c;
    }
}

// W_eff = mask ? codebook[idx] : W_float ; write bf16 normal + transposed copies
__global__ __launch_bounds__(256) void merge_kernel(const float* __restrict__ wf,
                                                    const int* __restrict__ idx,
                                                    const unsigned char* __restrict__ mask,
                                                    const float* __restrict__ cb,
                                                    bf16* __restrict__ out,
                                                    bf16* __restrict__ outT,
                                                    int rows, int cols) {
    size_t e = (size_t)blockIdx.x * blockDim.x + threadIdx.x;
    size_t n = (size_t)rows * cols;
    if (e < n) {
        int r  = (int)(e / cols);
        int cc = (int)(e % cols);
        float v = mask[e] ? cb[idx[e]] : wf[e];
        bf16 b = (bf16)v;
        out[e] = b;
        outT[(size_t)cc * rows + r] = b;
    }
}

// ---------------------------------------------------------------------------
// Tiled bf16 WMMA GEMM:  C[M,N] = A[M,K] * B[K,N], with B passed TRANSPOSED
// (Bt[N,K] row-major) so A-tile and B-tile staging are identical contiguous
// patterns -> both staged with global_load_async_to_lds_b128 (ASYNCcnt).
//   MODE 0: c19 activation, store bf16        (h)
//   MODE 1: bias, store f32 AND bf16          (z)
//   MODE 2: bias, store bf16                  (t)
//   MODE 3: bias, store f32                   (recon)
// Block: 256 threads (8 waves). Block tile 128x128, K-stage 64 (2 WMMA k-steps).
// Waves 4(M) x 2(N); each wave: 32x64 = 2x4 WMMA 16x16 tiles, 16 WMMA/stage.
// Double-buffered LDS; per wave 8 async b128 issues per stage.
// ---------------------------------------------------------------------------
template <int MODE>
__global__ __launch_bounds__(256) void gemm_bf16_wmma(
        const bf16* __restrict__ A, const bf16* __restrict__ Bt,
        int M, int N, int K,
        float* __restrict__ outF, bf16* __restrict__ outB,
        const float* __restrict__ bias,
        const float* __restrict__ rhoA, const float* __restrict__ qA,
        const float* __restrict__ icA) {

    constexpr int BM = 128, BN = 128, BK = 64;
    constexpr int PAD = 8, LDSS = BK + PAD;   // 72 elems = 144 B row stride (16B-aligned)

    __shared__ __align__(16) bf16 sA[2][BM * LDSS];
    __shared__ __align__(16) bf16 sB[2][BN * LDSS];

    const int tid  = threadIdx.x;
    const int lane = tid & 31;
    const int wave = tid >> 5;
    const int wm   = wave >> 1;      // 0..3  -> M offset wm*32
    const int wn   = wave & 1;       // 0..1  -> N offset wn*64
    const int bm   = blockIdx.y * BM;
    const int bn   = blockIdx.x * BN;

    const bf16* Abase = A  + (size_t)bm * K;
    const bf16* Bbase = Bt + (size_t)bn * K;

    const uint32_t aL[2] = { lds_addr_of(sA[0]), lds_addr_of(sA[1]) };
    const uint32_t bL[2] = { lds_addr_of(sB[0]), lds_addr_of(sB[1]) };

    // staging geometry: 128 rows x 64 elems = 1024 chunks of 8 bf16 (16B) per tile;
    // 8 waves x 4 issues x 32 lanes = 1024.
    int srow[4], skc[4];
    uint32_t sgoff[4], sloff[4];
#pragma unroll
    for (int s = 0; s < 4; ++s) {
        int chunk = (wave * 4 + s) * 32 + lane;          // 0..1023
        srow[s] = chunk >> 3;
        skc[s]  = (chunk & 7) * 8;
        sgoff[s] = (uint32_t)(srow[s] * K + skc[s]) * 2u;
        sloff[s] = (uint32_t)(srow[s] * LDSS + skc[s]) * 2u;
    }

    v8f acc[2][4];
#pragma unroll
    for (int i = 0; i < 2; ++i)
#pragma unroll
        for (int j = 0; j < 4; ++j)
#pragma unroll
            for (int r = 0; r < 8; ++r) acc[i][j][r] = 0.0f;

    // fragment addressing (ISA 16-bit WMMA VGPR layouts, wave32)
    const int a_row    = wm * 32 + (lane & 15);
    const int a_kbase  = (lane >> 4) * 8;     // lanes 0-15: K 0..7/16..23 ; 16-31: 8..15/24..31
    const int b_col    = wn * 64 + (lane & 15);
    const int b_kbase  = (lane >> 4) * 8;     // lanes 0-15: K 0..7/16..23 ; 16-31: 8..15/24..31

    const int nt = K / BK;

    // ---- prologue: stage tile 0 ----
#pragma unroll
    for (int s = 0; s < 4; ++s) {
        async_copy_b128(aL[0] + sloff[s], sgoff[s], Abase);
        async_copy_b128(bL[0] + sloff[s], sgoff[s], Bbase);
    }

    for (int it = 0; it < nt; ++it) {
        const int buf = it & 1;

        // stage next tile into the other buffer
        if (it + 1 < nt) {
            const bf16* an = Abase + (it + 1) * BK;
            const bf16* bnx = Bbase + (it + 1) * BK;
#pragma unroll
            for (int s = 0; s < 4; ++s) {
                async_copy_b128(aL[buf ^ 1] + sloff[s], sgoff[s], an);
                async_copy_b128(bL[buf ^ 1] + sloff[s], sgoff[s], bnx);
            }
            wait_asynccnt<8>();   // current tile's 8 copies complete (in-order)
        } else {
            wait_asynccnt<0>();
        }
        // L2 prefetch two stages ahead
        if (it + 2 < nt) {
            __builtin_prefetch(Abase + (size_t)(tid >> 1) * K + (it + 2) * BK + (tid & 1) * 32, 0, 1);
            __builtin_prefetch(Bbase + (size_t)(tid >> 1) * K + (it + 2) * BK + (tid & 1) * 32, 0, 1);
        }
        __syncthreads();

        const bf16* cAp = sA[buf];
        const bf16* cBp = sB[buf];
#pragma unroll
        for (int kk = 0; kk < 2; ++kk) {          // two WMMA k-steps per stage
            const int ko = kk * 32;
            v16bf afrag[2];
#pragma unroll
            for (int i = 0; i < 2; ++i) {
                const bf16* base = cAp + (a_row + i * 16) * LDSS + ko + a_kbase;
                v8bf lo = *(const v8bf*)(base);
                v8bf hi = *(const v8bf*)(base + 16);
                afrag[i] = __builtin_shufflevector(lo, hi, 0, 1, 2, 3, 4, 5, 6, 7,
                                                   8, 9, 10, 11, 12, 13, 14, 15);
            }
            v16bf bfrag[4];
#pragma unroll
            for (int j = 0; j < 4; ++j) {
                const bf16* base = cBp + (b_col + j * 16) * LDSS + ko + b_kbase;
                v8bf lo = *(const v8bf*)(base);
                v8bf hi = *(const v8bf*)(base + 16);
                bfrag[j] = __builtin_shufflevector(lo, hi, 0, 1, 2, 3, 4, 5, 6, 7,
                                                   8, 9, 10, 11, 12, 13, 14, 15);
            }
#pragma unroll
            for (int i = 0; i < 2; ++i)
#pragma unroll
                for (int j = 0; j < 4; ++j)
                    acc[i][j] = __builtin_amdgcn_wmma_f32_16x16x32_bf16(
                        false, afrag[i], false, bfrag[j],
                        (short)0, acc[i][j], false, false);
        }
        __syncthreads();
    }

    // ---- epilogue ----
#pragma unroll
    for (int i = 0; i < 2; ++i) {
#pragma unroll
        for (int j = 0; j < 4; ++j) {
            const int n     = bn + wn * 64 + j * 16 + (lane & 15);
            const int mbase = bm + wm * 32 + i * 16 + ((lane >> 4) << 3);
            const float bv = bias[n];
            float rho = 0.0f, q = 0.0f, ic = 0.0f;
            if (MODE == 0) { rho = rhoA[n]; q = qA[n]; ic = icA[n]; }
#pragma unroll
            for (int r = 0; r < 8; ++r) {
                const int m = mbase + r;
                float v = acc[i][j][r] + bv;
                if (MODE == 0) {
                    v = rho * v + q * fast_tanh(v * ic);
                    outB[(size_t)m * N + n] = (bf16)v;
                } else if (MODE == 1) {
                    outF[(size_t)m * N + n] = v;
                    outB[(size_t)m * N + n] = (bf16)v;
                } else if (MODE == 2) {
                    outB[(size_t)m * N + n] = (bf16)v;
                } else {
                    outF[(size_t)m * N + n] = v;
                }
            }
        }
    }
}

// ---------------------------------------------------------------------------
// Host launch
// ---------------------------------------------------------------------------
extern "C" void kernel_launch(void* const* d_in, const int* in_sizes, int n_in,
                              void* d_out, int out_size, void* d_ws, size_t ws_size,
                              hipStream_t stream) {
    const float* x    = (const float*)d_in[0];
    const float* cb1  = (const float*)d_in[1];
    const float* cb2  = (const float*)d_in[2];
    const float* w1f  = (const float*)d_in[3];
    const float* w2f  = (const float*)d_in[4];
    const float* b1   = (const float*)d_in[5];
    const float* b2   = (const float*)d_in[6];
    const float* db1  = (const float*)d_in[7];
    const float* db2  = (const float*)d_in[8];
    const float* craw = (const float*)d_in[9];
    const float* rraw = (const float*)d_in[10];
    const int*   w1i  = (const int*)d_in[11];
    const int*   w2i  = (const int*)d_in[12];
    const unsigned char* m1 = (const unsigned char*)d_in[13];
    const unsigned char* m2 = (const unsigned char*)d_in[14];

    char* ws = (char*)d_ws;
    size_t off = 0;
    auto carve = [&](size_t bytes) -> void* {
        void* p = ws + off;
        off += (bytes + 255) & ~(size_t)255;
        return p;
    };
    bf16*  xb   = (bf16*) carve((size_t)BB * IN_DIM * 2);
    bf16*  w1e  = (bf16*) carve((size_t)IN_DIM * H_DIM * 2);   // [IN, H]
    bf16*  w1et = (bf16*) carve((size_t)IN_DIM * H_DIM * 2);   // [H, IN]
    bf16*  w2e  = (bf16*) carve((size_t)H_DIM * OUT_DIM * 2);  // [H, OUT]
    bf16*  w2et = (bf16*) carve((size_t)H_DIM * OUT_DIM * 2);  // [OUT, H]
    bf16*  hb   = (bf16*) carve((size_t)BB * H_DIM * 2);
    bf16*  zb   = (bf16*) carve((size_t)BB * OUT_DIM * 2);
    bf16*  tb   = (bf16*) carve((size_t)BB * H_DIM * 2);
    float* rhoA = (float*)carve((size_t)H_DIM * 4);
    float* qA   = (float*)carve((size_t)H_DIM * 4);
    float* icA  = (float*)carve((size_t)H_DIM * 4);

    float* recon = (float*)d_out;                                  // [B, IN_DIM]
    float* zout  = (float*)d_out + (size_t)BB * IN_DIM;            // [B, OUT_DIM]

    // ---- prep ----
    {
        size_t n = (size_t)BB * IN_DIM;
        f2bf_kernel<<<dim3((unsigned)((n + 255) / 256)), 256, 0, stream>>>(x, xb, n);
    }
    act_params_kernel<<<dim3((H_DIM + 255) / 256), 256, 0, stream>>>(
        craw, rraw, rhoA, qA, icA, H_DIM);
    {
        size_t n = (size_t)IN_DIM * H_DIM;
        merge_kernel<<<dim3((unsigned)((n + 255) / 256)), 256, 0, stream>>>(
            w1f, w1i, m1, cb1, w1e, w1et, IN_DIM, H_DIM);
    }
    {
        size_t n = (size_t)H_DIM * OUT_DIM;
        merge_kernel<<<dim3((unsigned)((n + 255) / 256)), 256, 0, stream>>>(
            w2f, w2i, m2, cb2, w2e, w2et, H_DIM, OUT_DIM);
    }

    // ---- GEMM chain (B operand passed in transposed, N-major form) ----
    // h = c19(x @ W1_eff + b1)      A=[B,IN]   Bt=w1et:[H,IN]
    gemm_bf16_wmma<0><<<dim3(H_DIM / 128, BB / 128), 256, 0, stream>>>(
        xb, w1et, BB, H_DIM, IN_DIM, nullptr, hb, b1, rhoA, qA, icA);
    // z = h @ W2_eff + b2           A=[B,H]    Bt=w2et:[OUT,H]
    gemm_bf16_wmma<1><<<dim3(OUT_DIM / 128, BB / 128), 256, 0, stream>>>(
        hb, w2et, BB, OUT_DIM, H_DIM, zout, zb, b2, nullptr, nullptr, nullptr);
    // t = z @ W2_eff^T + db1        A=[B,OUT]  Bt=w2e:[H,OUT]
    gemm_bf16_wmma<2><<<dim3(H_DIM / 128, BB / 128), 256, 0, stream>>>(
        zb, w2e, BB, H_DIM, OUT_DIM, nullptr, tb, db1, nullptr, nullptr, nullptr);
    // recon = t @ W1_eff^T + db2    A=[B,H]    Bt=w1e:[IN,H]
    gemm_bf16_wmma<3><<<dim3(IN_DIM / 128, BB / 128), 256, 0, stream>>>(
        tb, w1e, BB, IN_DIM, H_DIM, recon, nullptr, db2, nullptr, nullptr, nullptr);
}
